// LlamaForCausalLM_36928128811668
// MI455X (gfx1250) — matrix operand order repaired and checked by reading, not compiled, not run on hardware
//
#include <hip/hip_runtime.h>
#include <hip/hip_bf16.h>

// ---------------------------------------------------------------------------
// Llama-2-ish forward (B=1, S=2048, D=2048, H=16, HK=4, HD=128, I=5632, L=2,
// V=32064) for gfx1250 (MI455X). All heavy GEMMs + attention matmuls run on
// v_wmma_f32_16x16x32_bf16 (fp32 operands converted to bf16 on the fly, fp32
// accumulate). Softmax / norms / residuals stay fp32.
//
// Round-3 change: GEMM B (weight) tiles are now staged through LDS:
// coalesced float4 global loads -> packed v_cvt_pk_bf16_f32 -> transposed
// K-major LDS layout (stride 20 DWORDs: 16B-aligned ds_load_b128 fragments,
// conflict-spread banks), double-buffered with one barrier per K-step. This
// removes the serialized 16x global_load_b32 + s_wait chain per B fragment
// and the 4x cross-wave load redundancy that throttled WMMA issue.
// ---------------------------------------------------------------------------

#define Bsz 1
#define Ssz 2048
#define Dsz 2048
#define Hq  16
#define Hkv 4
#define HDs 128
#define Isz 5632
#define Lnum 2
#define Vsz 32064
#define QKVW ((Hq + 2 * Hkv) * HDs)   // 3072
#define ATT_SCALE 0.08838834764831845f // 1/sqrt(128)

typedef __attribute__((ext_vector_type(16))) __bf16 bf16x16;
typedef __attribute__((ext_vector_type(8)))  float  f32x8;

// fp32 -> bf16, native fptrunc (round-to-nearest-even); backend emits
// v_cvt_pk_bf16_f32 for pairs.
__device__ __forceinline__ __bf16 cvt_bf16(float f) { return (__bf16)f; }

// Pack two fp32 into one DWORD of two bf16 (lo = first arg).
__device__ __forceinline__ unsigned pack2_bf16(float lo, float hi) {
  union { __bf16 h[2]; unsigned u; } z;
  z.h[0] = (__bf16)lo; z.h[1] = (__bf16)hi;
  return z.u;
}

// A-matrix 16x32 bf16 tile loader:
//   lane half hi: element e -> K = e + 16*hi
//   => one contiguous 16-float run at k0+16*hi -> 4x float4.
__device__ __forceinline__ bf16x16 load_a16x32(const float* __restrict__ A,
                                               int lda, int row, int k0, int hi) {
  const float* p = A + (size_t)row * lda + k0 + hi * 16;
  float4 f0 = *(const float4*)(p + 0);
  float4 f1 = *(const float4*)(p + 4);
  float4 f2 = *(const float4*)(p + 8);
  float4 f3 = *(const float4*)(p + 12);
  bf16x16 a;
  a[0]  = cvt_bf16(f0.x); a[1]  = cvt_bf16(f0.y); a[2]  = cvt_bf16(f0.z); a[3]  = cvt_bf16(f0.w);
  a[4]  = cvt_bf16(f1.x); a[5]  = cvt_bf16(f1.y); a[6]  = cvt_bf16(f1.z); a[7]  = cvt_bf16(f1.w);
  a[8]  = cvt_bf16(f2.x); a[9]  = cvt_bf16(f2.y); a[10] = cvt_bf16(f2.z); a[11] = cvt_bf16(f2.w);
  a[12] = cvt_bf16(f3.x); a[13] = cvt_bf16(f3.y); a[14] = cvt_bf16(f3.z); a[15] = cvt_bf16(f3.w);
  return a;
}

__device__ __forceinline__ void store_c16x16(float* __restrict__ C, int ldc,
                                             int row0, int col0, int l16, int hi,
                                             f32x8 acc, const float* __restrict__ Res) {
  int col = col0 + l16;
#pragma unroll
  for (int r = 0; r < 8; ++r) {
    int row = row0 + r + hi * 8;                // C layout: VGPR r -> M = r + 8*hi
    float v = acc[r];
    if (Res) v += Res[(size_t)row * ldc + col];
    C[(size_t)row * ldc + col] = v;
  }
}

// LDS B-tile geometry: [2 buffers][64 cols][stride 20 DWORDs].
// Word kp of column c holds bf16(K=2kp) | bf16(K=2kp+1)<<16, so the 8
// consecutive DWORDs at [c][8*hi] are exactly the bf16x16 WMMA B fragment
// (element e -> K = e + 16*hi).
#define BSTRIDE 20

__device__ __forceinline__ bf16x16 lds_b_frag(const unsigned* __restrict__ base,
                                              int col, int hi) {
  const unsigned* p = base + col * BSTRIDE + hi * 8;  // 16B aligned (20*4, 8*4)
  union { uint4 q[2]; bf16x16 v; } u;
  u.q[0] = *(const uint4*)(p + 0);   // ds_load_b128
  u.q[1] = *(const uint4*)(p + 4);   // ds_load_b128
  return u.v;
}

// ---------------------------------------------------------------------------
// Generic GEMM:  C[M,N] = (Res ? Res : 0) + A[M,K] @ B[K,N]   (row-major fp32)
// Grid: (N/64, M/128), 256 threads = 8 waves, wave tile 32x32 (2x2 WMMA).
// Requires M%128==0, N%64==0, K%32==0 (true for every GEMM in this model).
// ---------------------------------------------------------------------------
__global__ __launch_bounds__(256, 2)
void gemm_bf16_wmma(const float* __restrict__ A, const float* __restrict__ B,
                    const float* __restrict__ Res, float* __restrict__ C,
                    int M, int N, int K) {
  __shared__ alignas(16) unsigned bsh[2][64 * BSTRIDE];

  const int t    = threadIdx.x;
  const int lane = t & 31;
  const int wave = t >> 5;
  const int l16  = lane & 15;
  const int hi   = lane >> 4;
  const int wm   = wave >> 1;   // 0..3
  const int wn   = wave & 1;    // 0..1
  const int row0 = blockIdx.y * 128 + wm * 32;
  const int col0 = blockIdx.x * 64  + wn * 32;   // global col of wave tile
  const int colL = wn * 32;                      // col within LDS tile

  // Cooperative B-tile fill: thread handles k-pair kp (2 rows) x 4 cols.
  const int kp = t >> 4;          // 0..15 -> K rows 2kp, 2kp+1
  const int cg = (t & 15) * 4;    // 4 consecutive cols
  const float* Bt = B + (size_t)blockIdx.x * 64 + cg;

  f32x8 acc00 = {}, acc01 = {}, acc10 = {}, acc11 = {};

  // Prologue fill of buffer 0.
  {
    const float* r0 = Bt + (size_t)(2 * kp) * N;
    const float* r1 = r0 + N;
    float4 fa = *(const float4*)r0, fb = *(const float4*)r1;
    unsigned* d = &bsh[0][0];
    d[(cg + 0) * BSTRIDE + kp] = pack2_bf16(fa.x, fb.x);
    d[(cg + 1) * BSTRIDE + kp] = pack2_bf16(fa.y, fb.y);
    d[(cg + 2) * BSTRIDE + kp] = pack2_bf16(fa.z, fb.z);
    d[(cg + 3) * BSTRIDE + kp] = pack2_bf16(fa.w, fb.w);
  }
  __syncthreads();

  for (int k0 = 0; k0 < K; k0 += 32) {
    const int p = (k0 >> 5) & 1;

    // A fragments straight from global (b128 per lane, clause-friendly).
    bf16x16 a0 = load_a16x32(A, K, row0 + l16,      k0, hi);
    bf16x16 a1 = load_a16x32(A, K, row0 + 16 + l16, k0, hi);
    // B fragments from LDS (2x ds_load_b128 each, bit-exact layout).
    bf16x16 b0 = lds_b_frag(&bsh[p][0], colL + l16,      hi);
    bf16x16 b1 = lds_b_frag(&bsh[p][0], colL + 16 + l16, hi);

    // Overlap: fill the other buffer with the next K tile.
    if (k0 + 32 < K) {
      __builtin_prefetch(A + (size_t)(row0 + l16) * K + k0 + 64, 0, 0);  // global_prefetch_b8
      const float* r0 = Bt + (size_t)(k0 + 32 + 2 * kp) * N;
      const float* r1 = r0 + N;
      float4 fa = *(const float4*)r0, fb = *(const float4*)r1;
      unsigned* d = &bsh[p ^ 1][0];
      d[(cg + 0) * BSTRIDE + kp] = pack2_bf16(fa.x, fb.x);
      d[(cg + 1) * BSTRIDE + kp] = pack2_bf16(fa.y, fb.y);
      d[(cg + 2) * BSTRIDE + kp] = pack2_bf16(fa.z, fb.z);
      d[(cg + 3) * BSTRIDE + kp] = pack2_bf16(fa.w, fb.w);
    }
    __syncthreads();

    acc00 = __builtin_amdgcn_wmma_f32_16x16x32_bf16(false, a0, false, b0, (short)0, acc00, false, false);
    acc01 = __builtin_amdgcn_wmma_f32_16x16x32_bf16(false, a0, false, b1, (short)0, acc01, false, false);
    acc10 = __builtin_amdgcn_wmma_f32_16x16x32_bf16(false, a1, false, b0, (short)0, acc10, false, false);
    acc11 = __builtin_amdgcn_wmma_f32_16x16x32_bf16(false, a1, false, b1, (short)0, acc11, false, false);
  }
  store_c16x16(C, N, row0,      col0,      l16, hi, acc00, Res);
  store_c16x16(C, N, row0,      col0 + 16, l16, hi, acc01, Res);
  store_c16x16(C, N, row0 + 16, col0,      l16, hi, acc10, Res);
  store_c16x16(C, N, row0 + 16, col0 + 16, l16, hi, acc11, Res);
}

// ---------------------------------------------------------------------------
// Flash attention (causal, GQA 4:1). One wave per block; block = (q-tile of 16,
// head). QK^T and P@V run on WMMA; P transposes C-layout -> A-layout via LDS.
// ---------------------------------------------------------------------------
__global__ __launch_bounds__(32)
void flash_attn(const float* __restrict__ qkv, float* __restrict__ out) {
  const int lane = threadIdx.x & 31;
  const int l16  = lane & 15;
  const int hi   = lane >> 4;
  const int q0   = blockIdx.x * 16;
  const int h    = blockIdx.y;
  const int kvh  = h >> 2;                    // H/HK = 4
  const int kOff = (Hq + kvh) * HDs;          // K head column offset in qkv row
  const int vOff = (Hq + Hkv + kvh) * HDs;    // V head column offset

  __shared__ float pl[16][32];

  // Q tile: 16 queries x 128 dims as 4 A-matrices (K-chunks of 32).
  bf16x16 aq[4];
#pragma unroll
  for (int c = 0; c < 4; ++c)
    aq[c] = load_a16x32(qkv + h * HDs, QKVW, q0 + l16, c * 32, hi);

  f32x8 o[8] = {};               // 16 x 128 output accumulator (8 N-chunks)
  float mrow[8], lrow[8], alphas[8];
#pragma unroll
  for (int r = 0; r < 8; ++r) { mrow[r] = -3.0e38f; lrow[r] = 0.0f; }

  for (int k0 = 0; k0 < q0 + 16; k0 += 32) {   // causal: keys <= q0+15
    // ---- scores S = Q @ K^T : two 16x16 f32 C-tiles (keys k0.. and k0+16..)
    f32x8 s0 = {}, s1 = {};
#pragma unroll
    for (int c = 0; c < 4; ++c) {
      // B(K=hd, N=key): per lane, 16 contiguous floats along hd.
      const float* p0 = qkv + (size_t)(k0 + l16) * QKVW + kOff + c * 32 + hi * 16;
      const float* p1 = qkv + (size_t)(k0 + 16 + l16) * QKVW + kOff + c * 32 + hi * 16;
      float4 g0 = *(const float4*)(p0 + 0), g1 = *(const float4*)(p0 + 4);
      float4 g2 = *(const float4*)(p0 + 8), g3 = *(const float4*)(p0 + 12);
      float4 u0 = *(const float4*)(p1 + 0), u1 = *(const float4*)(p1 + 4);
      float4 u2 = *(const float4*)(p1 + 8), u3 = *(const float4*)(p1 + 12);
      bf16x16 bk0, bk1;
      bk0[0]  = cvt_bf16(g0.x); bk0[1]  = cvt_bf16(g0.y); bk0[2]  = cvt_bf16(g0.z); bk0[3]  = cvt_bf16(g0.w);
      bk0[4]  = cvt_bf16(g1.x); bk0[5]  = cvt_bf16(g1.y); bk0[6]  = cvt_bf16(g1.z); bk0[7]  = cvt_bf16(g1.w);
      bk0[8]  = cvt_bf16(g2.x); bk0[9]  = cvt_bf16(g2.y); bk0[10] = cvt_bf16(g2.z); bk0[11] = cvt_bf16(g2.w);
      bk0[12] = cvt_bf16(g3.x); bk0[13] = cvt_bf16(g3.y); bk0[14] = cvt_bf16(g3.z); bk0[15] = cvt_bf16(g3.w);
      bk1[0]  = cvt_bf16(u0.x); bk1[1]  = cvt_bf16(u0.y); bk1[2]  = cvt_bf16(u0.z); bk1[3]  = cvt_bf16(u0.w);
      bk1[4]  = cvt_bf16(u1.x); bk1[5]  = cvt_bf16(u1.y); bk1[6]  = cvt_bf16(u1.z); bk1[7]  = cvt_bf16(u1.w);
      bk1[8]  = cvt_bf16(u2.x); bk1[9]  = cvt_bf16(u2.y); bk1[10] = cvt_bf16(u2.z); bk1[11] = cvt_bf16(u2.w);
      bk1[12] = cvt_bf16(u3.x); bk1[13] = cvt_bf16(u3.y); bk1[14] = cvt_bf16(u3.z); bk1[15] = cvt_bf16(u3.w);
      s0 = __builtin_amdgcn_wmma_f32_16x16x32_bf16(false, aq[c], false, bk0, (short)0, s0, false, false);
      s1 = __builtin_amdgcn_wmma_f32_16x16x32_bf16(false, aq[c], false, bk1, (short)0, s1, false, false);
    }

    // ---- online softmax update (row = r + 8*hi; cols striped over 16 lanes)
#pragma unroll
    for (int r = 0; r < 8; ++r) {
      const int qrow = q0 + r + 8 * hi;
      float v0 = s0[r] * ATT_SCALE; if (k0 + l16 > qrow)      v0 = -1.0e30f;
      float v1 = s1[r] * ATT_SCALE; if (k0 + 16 + l16 > qrow) v1 = -1.0e30f;
      float t = fmaxf(v0, v1);
#pragma unroll
      for (int off = 8; off >= 1; off >>= 1) t = fmaxf(t, __shfl_xor(t, off, 32));
      const float mnew  = fmaxf(mrow[r], t);
      const float alpha = __expf(mrow[r] - mnew);
      const float p0 = __expf(v0 - mnew);
      const float p1 = __expf(v1 - mnew);
      float rs = p0 + p1;
#pragma unroll
      for (int off = 8; off >= 1; off >>= 1) rs += __shfl_xor(rs, off, 32);
      lrow[r]   = lrow[r] * alpha + rs;
      mrow[r]   = mnew;
      alphas[r] = alpha;
      pl[r + 8 * hi][l16]      = p0;   // stage P for layout transpose
      pl[r + 8 * hi][16 + l16] = p1;
    }
#pragma unroll
    for (int j = 0; j < 8; ++j)
#pragma unroll
      for (int r = 0; r < 8; ++r) o[j][r] *= alphas[r];

    asm volatile("s_wait_dscnt 0" ::: "memory");

    // P (16x32) back out in A-matrix layout, bf16.
    bf16x16 ap;
#pragma unroll
    for (int e = 0; e < 16; ++e) {
      const int kk = e + hi * 16;
      ap[e] = cvt_bf16(pl[l16][kk]);
    }

    // ---- O += P @ V  (V tile: B(K=key, N=hd), strided loads)
#pragma unroll
    for (int j = 0; j < 8; ++j) {
      bf16x16 bv;
#pragma unroll
      for (int e = 0; e < 16; ++e) {
        const int key = k0 + e + 16 * hi;
        bv[e] = cvt_bf16(qkv[(size_t)key * QKVW + vOff + j * 16 + l16]);
      }
      o[j] = __builtin_amdgcn_wmma_f32_16x16x32_bf16(false, ap, false, bv, (short)0, o[j], false, false);
    }
  }

  // ---- normalize and write O [16 x 128] into out[s, h*128 + :]
#pragma unroll
  for (int r = 0; r < 8; ++r) {
    const float inv = 1.0f / lrow[r];
    const size_t row = (size_t)(q0 + r + 8 * hi);
#pragma unroll
    for (int j = 0; j < 8; ++j)
      out[row * (Hq * HDs) + h * HDs + j * 16 + l16] = o[j][r] * inv;
  }
}

// ---------------------------------------------------------------------------
// Elementwise kernels
// ---------------------------------------------------------------------------
__global__ __launch_bounds__(256)
void embed_gather(const int* __restrict__ ids, const float* __restrict__ embed,
                  float* __restrict__ x) {
  const int s = blockIdx.x, t = threadIdx.x;
  const int id = ids[s];
  const float4* src = (const float4*)(embed + (size_t)id * Dsz);
  float4* dst = (float4*)(x + (size_t)s * Dsz);
  dst[t]       = src[t];
  dst[t + 256] = src[t + 256];
}

__global__ __launch_bounds__(256)
void rmsnorm(const float* __restrict__ x, const float* __restrict__ w,
             float* __restrict__ out) {
  __shared__ float sred[8];
  const int row = blockIdx.x, t = threadIdx.x;
  const float* base = x + (size_t)row * Dsz;
  float ss = 0.0f;
#pragma unroll
  for (int i = t; i < Dsz; i += 256) { float v = base[i]; ss += v * v; }
#pragma unroll
  for (int off = 16; off >= 1; off >>= 1) ss += __shfl_xor(ss, off, 32);
  if ((t & 31) == 0) sred[t >> 5] = ss;
  __syncthreads();
  float tot = 0.0f;
#pragma unroll
  for (int wv = 0; wv < 8; ++wv) tot += sred[wv];
  const float scale = rsqrtf(tot * (1.0f / Dsz) + 1e-5f);
#pragma unroll
  for (int i = t; i < Dsz; i += 256)
    out[(size_t)row * Dsz + i] = base[i] * scale * w[i];
}

// NeoX rotate-half RoPE, applied in place to Q heads (0..15) and K heads
// (16..19) of the packed qkv buffer.
__global__ __launch_bounds__(64)
void rope_kernel(float* __restrict__ qkv, const int* __restrict__ pos) {
  const int s = blockIdx.x, head = blockIdx.y, j = threadIdx.x;  // j in [0,64)
  const float p = (float)pos[s];
  const float freq = __expf(-(2.0f * (float)j / (float)HDs) * 9.210340372f); // ln(1e4)
  const float ang = p * freq;
  const float c = __cosf(ang), sn = __sinf(ang);
  float* base = qkv + (size_t)s * QKVW + head * HDs;
  const float x1 = base[j], x2 = base[j + 64];
  base[j]      = x1 * c - x2 * sn;
  base[j + 64] = x2 * c + x1 * sn;
}

__global__ __launch_bounds__(256)
void swiglu(const float* __restrict__ gu, float* __restrict__ hm) {
  const int idx = blockIdx.x * 256 + threadIdx.x;      // over S*I
  const int s = idx / Isz, i = idx - s * Isz;
  const float g = gu[(size_t)s * (2 * Isz) + i];
  const float u = gu[(size_t)s * (2 * Isz) + Isz + i];
  hm[idx] = (g / (1.0f + __expf(-g))) * u;
}

// ---------------------------------------------------------------------------
// Host-side orchestration
// ---------------------------------------------------------------------------
extern "C" void kernel_launch(void* const* d_in, const int* in_sizes, int n_in,
                              void* d_out, int out_size, void* d_ws, size_t ws_size,
                              hipStream_t stream) {
  const int*   ids   = (const int*)d_in[0];
  const int*   pos   = (const int*)d_in[1];
  const float* embed = (const float*)d_in[2];
  const float* wqkv  = (const float*)d_in[3];
  const float* wo    = (const float*)d_in[4];
  const float* wgu   = (const float*)d_in[5];
  const float* wdn   = (const float*)d_in[6];
  const float* ln1   = (const float*)d_in[7];
  const float* ln2   = (const float*)d_in[8];
  const float* normw = (const float*)d_in[9];
  const float* lmh   = (const float*)d_in[10];
  float* logits = (float*)d_out;

  const size_t SD = (size_t)Ssz * Dsz;
  float* x   = (float*)d_ws;            // [S,D] residual stream
  float* h   = x   + SD;                // [S,D] normed activations
  float* qkv = h   + SD;                // [S,3072]
  float* ao  = qkv + (size_t)Ssz * QKVW;// [S,2048] attention output
  float* gu  = ao  + SD;                // [S,11264]
  float* hm  = gu  + (size_t)Ssz * 2 * Isz; // [S,5632]

  embed_gather<<<Ssz, 256, 0, stream>>>(ids, embed, x);

  for (int l = 0; l < Lnum; ++l) {
    const float* wqkv_l = wqkv + (size_t)l * Dsz * QKVW;
    const float* wo_l   = wo   + (size_t)l * (Hq * HDs) * Dsz;
    const float* wgu_l  = wgu  + (size_t)l * Dsz * (2 * Isz);
    const float* wdn_l  = wdn  + (size_t)l * Isz * Dsz;

    // --- attention block ---
    rmsnorm<<<Ssz, 256, 0, stream>>>(x, ln1 + (size_t)l * Dsz, h);
    gemm_bf16_wmma<<<dim3(QKVW / 64, Ssz / 128), 256, 0, stream>>>(
        h, wqkv_l, nullptr, qkv, Ssz, QKVW, Dsz);
    rope_kernel<<<dim3(Ssz, Hq + Hkv), 64, 0, stream>>>(qkv, pos);
    flash_attn<<<dim3(Ssz / 16, Hq), 32, 0, stream>>>(qkv, ao);
    gemm_bf16_wmma<<<dim3(Dsz / 64, Ssz / 128), 256, 0, stream>>>(
        ao, wo_l, x, x, Ssz, Dsz, Hq * HDs);

    // --- SwiGLU MLP block ---
    rmsnorm<<<Ssz, 256, 0, stream>>>(x, ln2 + (size_t)l * Dsz, h);
    gemm_bf16_wmma<<<dim3((2 * Isz) / 64, Ssz / 128), 256, 0, stream>>>(
        h, wgu_l, nullptr, gu, Ssz, 2 * Isz, Dsz);
    swiglu<<<(Ssz * Isz) / 256, 256, 0, stream>>>(gu, hm);
    gemm_bf16_wmma<<<dim3(Dsz / 64, Ssz / 128), 256, 0, stream>>>(
        hm, wdn_l, x, x, Ssz, Dsz, Isz);
  }

  // --- final norm + lm_head ---
  rmsnorm<<<Ssz, 256, 0, stream>>>(x, normw, h);
  gemm_bf16_wmma<<<dim3(Vsz / 64, Ssz / 128), 256, 0, stream>>>(
      h, lmh, nullptr, logits, Ssz, Vsz, Dsz);
}
